// DeformableTransformerDecoderLayer_87445534146765
// MI455X (gfx1250) — matrix operand order repaired and checked by reading, not compiled, not run on hardware
//
#include <hip/hip_runtime.h>

// ---------------------------------------------------------------------------
// Deformable transformer decoder layer for MI455X (gfx1250).
// All GEMMs run on v_wmma_f32_16x16x32_bf16 (bf16 A/B, f32 accumulate).
// GEMM tiles are staged with CDNA5 async global->LDS DMA (ASYNCcnt).
// ---------------------------------------------------------------------------

typedef __attribute__((ext_vector_type(16))) __bf16 v16bf;
typedef __attribute__((ext_vector_type(8)))  float  v8f;

union BF16Frag {
    v16bf          v;
    unsigned int   u[8];
    unsigned short s[16];
};

__device__ __forceinline__ unsigned short f2bf(float f) {
    unsigned int u = __float_as_uint(f);
    u += 0x7fffu + ((u >> 16) & 1u);        // round-to-nearest-even
    return (unsigned short)(u >> 16);
}

static constexpr int NB   = 4;       // batch
static constexpr int LQ   = 900;     // queries
static constexpr int LEN  = 17821;   // sum h*w
static constexpr int M36  = NB * LQ; // 3600
static constexpr int MFEAT = NB * LEN; // 71284

// ---------------------------------------------------------------------------
// Generic WMMA GEMM:  C[M,N] = A[M,K] (bf16, row-major) * W[N,K]^T + bias
// block = 128 threads (4 waves), block tile 64x64, wave tile 16x64.
// Tiles staged via global_load_async_to_lds_b128 (no VGPR bounce).
// ---------------------------------------------------------------------------
__global__ __launch_bounds__(128)
void gemm_bf16_wmma(const unsigned short* __restrict__ A,
                    const unsigned short* __restrict__ W,
                    const float* __restrict__ bias,
                    float* __restrict__ Cf,
                    unsigned short* __restrict__ Cbf,
                    int M, int N, int K, int ldc, int relu)
{
    // 40-short row stride = 80B: every 16B async-DMA destination is 16B aligned
    __shared__ unsigned short As[64][40];
    __shared__ unsigned short Bs[64][40];

    const int tid  = threadIdx.x;
    const int wave = tid >> 5;
    const int lane = tid & 31;
    const int mloc = lane & 15;
    const int kh   = lane >> 4;
    const int m0 = blockIdx.x * 64;
    const int n0 = blockIdx.y * 64;

    v8f acc[4] = {};

    for (int k0 = 0; k0 < K; k0 += 32) {
        // ---- stage 64x32 bf16 of A and of W via async DMA (16B per op) ----
#pragma unroll
        for (int i = 0; i < 2; ++i) {
            int idx = tid + i * 128;          // 0..255
            int r   = idx >> 2;               // row 0..63
            int c8  = (idx & 3) * 8;          // short col 0/8/16/24
            int ar = m0 + r; ar = ar < M ? ar : M - 1;   // clamp: extra rows discarded later
            int wr = n0 + r; wr = wr < N ? wr : N - 1;
            const unsigned short* gA = A + (size_t)ar * K + k0 + c8;
            const unsigned short* gB = W + (size_t)wr * K + k0 + c8;
            unsigned la = (unsigned)(unsigned long long)(uintptr_t)&As[r][c8];
            unsigned lb = (unsigned)(unsigned long long)(uintptr_t)&Bs[r][c8];
            asm volatile("global_load_async_to_lds_b128 %0, %1, off"
                         :: "v"(la), "v"((unsigned long long)(uintptr_t)gA) : "memory");
            asm volatile("global_load_async_to_lds_b128 %0, %1, off"
                         :: "v"(lb), "v"((unsigned long long)(uintptr_t)gB) : "memory");
            if (k0 + 32 < K) {                 // prefetch next K tile into L2
                __builtin_prefetch(gA + 32, 0, 1);
                __builtin_prefetch(gB + 32, 0, 1);
            }
        }
        asm volatile("s_wait_asynccnt 0x0" ::: "memory");
        __syncthreads();

        // A fragment: 16x32, rows wave*16 .. wave*16+15
        BF16Frag a;
        {
            int am = wave * 16 + mloc;
#pragma unroll
            for (int j = 0; j < 8; ++j) {
                int kb = 16 * kh + 2 * j;
                a.u[j] = *(const unsigned int*)&As[am][kb];
            }
        }
#pragma unroll
        for (int t = 0; t < 4; ++t) {
            BF16Frag b;
            int bn = t * 16 + mloc;
#pragma unroll
            for (int j = 0; j < 8; ++j) {
                int kb = 16 * kh + 2 * j;
                b.u[j] = *(const unsigned int*)&Bs[bn][kb];
            }
            acc[t] = __builtin_amdgcn_wmma_f32_16x16x32_bf16(
                false, a.v, false, b.v, (short)0, acc[t], false, false);
        }
        __syncthreads();
    }

    // epilogue: C layout n = lane%16, row = r + 8*(lane/16)
#pragma unroll
    for (int t = 0; t < 4; ++t) {
        int col = n0 + t * 16 + mloc;
#pragma unroll
        for (int r = 0; r < 8; ++r) {
            int row = m0 + wave * 16 + r + 8 * kh;
            if (row < M && col < N) {
                float v = acc[t][r] + (bias ? bias[col] : 0.f);
                if (relu) v = fmaxf(v, 0.f);
                if (Cf)  Cf[(size_t)row * ldc + col] = v;
                if (Cbf) Cbf[(size_t)row * ldc + col] = f2bf(v);
            }
        }
    }
}

// ---------------------------------------------------------------------------
// Flash self-attention, one wave per (batch, head, 16-query tile).
// QKV: (N*LQ, 768) bf16, q|k|v at col offsets 0/256/512, head h -> cols h*32..
// ---------------------------------------------------------------------------
__global__ __launch_bounds__(32)
void mha_flash_wmma(const unsigned short* __restrict__ QKV,
                    unsigned short* __restrict__ Obf)
{
    __shared__ unsigned short Pl[16][36];

    const int lane = threadIdx.x;
    const int mloc = lane & 15;
    const int kh   = lane >> 4;
    const int qt = blockIdx.x, h = blockIdx.y, n = blockIdx.z;
    const int q0 = qt * 16;
    const float scale = 0.17677669529663687f;   // 1/sqrt(32)

    // Q fragment (A-layout 16x32)
    BF16Frag qf;
    {
        int q = q0 + mloc;
        int qc = q < LQ ? q : (LQ - 1);
        const unsigned short* qp = QKV + ((size_t)(n * LQ + qc)) * 768 + h * 32;
#pragma unroll
        for (int j = 0; j < 8; ++j) {
            int kb = 16 * kh + 2 * j;
            unsigned int u = *(const unsigned int*)(qp + kb);
            qf.u[j] = (q < LQ) ? u : 0u;
        }
    }

    v8f o0 = {}, o1 = {};
    float mi[8], li[8];
#pragma unroll
    for (int r = 0; r < 8; ++r) { mi[r] = -1e30f; li[r] = 0.f; }

    for (int k0 = 0; k0 < LQ; k0 += 32) {
        // S = Q * K^T  (two 16-key halves)
        v8f s01[2];
#pragma unroll
        for (int half = 0; half < 2; ++half) {
            BF16Frag bk;
            int key = k0 + half * 16 + mloc;
            int kc = key < LQ ? key : (LQ - 1);
            const unsigned short* kp = QKV + ((size_t)(n * LQ + kc)) * 768 + 256 + h * 32;
#pragma unroll
            for (int j = 0; j < 8; ++j) {
                unsigned int u = *(const unsigned int*)(kp + 16 * kh + 2 * j);
                bk.u[j] = (key < LQ) ? u : 0u;
            }
            v8f z = {};
            s01[half] = __builtin_amdgcn_wmma_f32_16x16x32_bf16(
                false, qf.v, false, bk.v, (short)0, z, false, false);
        }

        // Online softmax in C layout: lane col key, rows r + 8*kh
        float pv0[8], pv1[8];
        bool v0k = (k0 + mloc) < LQ;
        bool v1k = (k0 + 16 + mloc) < LQ;
#pragma unroll
        for (int r = 0; r < 8; ++r) {
            float s0 = v0k ? s01[0][r] * scale : -1e30f;
            float s1 = v1k ? s01[1][r] * scale : -1e30f;
            float rm = fmaxf(s0, s1);
#pragma unroll
            for (int d = 1; d < 16; d <<= 1) rm = fmaxf(rm, __shfl_xor(rm, d, 16));
            float mnew = fmaxf(mi[r], rm);
            float corr = __expf(mi[r] - mnew);
            float p0 = v0k ? __expf(s0 - mnew) : 0.f;
            float p1 = v1k ? __expf(s1 - mnew) : 0.f;
            float rs = p0 + p1;
#pragma unroll
            for (int d = 1; d < 16; d <<= 1) rs += __shfl_xor(rs, d, 16);
            li[r] = li[r] * corr + rs;
            mi[r] = mnew;
            o0[r] *= corr;
            o1[r] *= corr;
            pv0[r] = p0; pv1[r] = p1;
        }

        // P through LDS: C-layout -> A-layout transpose (per-wave, LDS in-order)
#pragma unroll
        for (int r = 0; r < 8; ++r) {
            int m = r + 8 * kh;
            Pl[m][mloc]      = f2bf(pv0[r]);
            Pl[m][16 + mloc] = f2bf(pv1[r]);
        }
        asm volatile("s_wait_dscnt 0" ::: "memory");
        BF16Frag pf;
#pragma unroll
        for (int j = 0; j < 8; ++j) {
            int kb = 16 * kh + 2 * j;
            pf.u[j] = *(const unsigned int*)&Pl[mloc][kb];
        }
        asm volatile("s_wait_dscnt 0" ::: "memory");

        // O += P * V, two dh halves
#pragma unroll
        for (int hf = 0; hf < 2; ++hf) {
            BF16Frag bv;
            int d = hf * 16 + mloc;
#pragma unroll
            for (int j = 0; j < 8; ++j) {
                int key0 = k0 + 16 * kh + 2 * j;
                int key1 = key0 + 1;
                unsigned short e0 = (key0 < LQ)
                    ? QKV[((size_t)(n * LQ + key0)) * 768 + 512 + h * 32 + d] : (unsigned short)0;
                unsigned short e1 = (key1 < LQ)
                    ? QKV[((size_t)(n * LQ + key1)) * 768 + 512 + h * 32 + d] : (unsigned short)0;
                bv.u[j] = (unsigned int)e0 | ((unsigned int)e1 << 16);
            }
            if (hf == 0)
                o0 = __builtin_amdgcn_wmma_f32_16x16x32_bf16(
                    false, pf.v, false, bv.v, (short)0, o0, false, false);
            else
                o1 = __builtin_amdgcn_wmma_f32_16x16x32_bf16(
                    false, pf.v, false, bv.v, (short)0, o1, false, false);
        }
    }

    // epilogue: out (N*LQ, 256) bf16
#pragma unroll
    for (int r = 0; r < 8; ++r) {
        int q = q0 + r + 8 * kh;
        if (q < LQ) {
            float inv = (li[r] > 0.f) ? 1.f / li[r] : 0.f;
            size_t row = (size_t)(n * LQ + q) * 256 + h * 32;
            Obf[row + mloc]      = f2bf(o0[r] * inv);
            Obf[row + 16 + mloc] = f2bf(o1[r] * inv);
        }
    }
}

// ---------------------------------------------------------------------------
// Residual + LayerNorm: y = LN(a + c) * g + b ; one row (D=256) per block.
// ---------------------------------------------------------------------------
__global__ __launch_bounds__(256)
void residual_ln(const float* __restrict__ a, const float* __restrict__ c,
                 const float* __restrict__ g, const float* __restrict__ b,
                 float* __restrict__ y, unsigned short* __restrict__ ybf)
{
    __shared__ float red[256];
    const int row = blockIdx.x, t = threadIdx.x;
    float v = a[(size_t)row * 256 + t] + c[(size_t)row * 256 + t];
    red[t] = v; __syncthreads();
    for (int s = 128; s > 0; s >>= 1) { if (t < s) red[t] += red[t + s]; __syncthreads(); }
    float mean = red[0] * (1.f / 256.f); __syncthreads();
    float d = v - mean;
    red[t] = d * d; __syncthreads();
    for (int s = 128; s > 0; s >>= 1) { if (t < s) red[t] += red[t + s]; __syncthreads(); }
    float var = red[0] * (1.f / 256.f);
    float o = d * rsqrtf(var + 1e-5f) * g[t] + b[t];
    y[(size_t)row * 256 + t] = o;
    if (ybf) ybf[(size_t)row * 256 + t] = f2bf(o);
}

// ---------------------------------------------------------------------------
// Elementwise helpers
// ---------------------------------------------------------------------------
__global__ void k_to_bf16(const float* __restrict__ a, unsigned short* __restrict__ o, size_t n)
{
    size_t i = (size_t)blockIdx.x * blockDim.x + threadIdx.x;
    if (i < n) o[i] = f2bf(a[i]);
}

__global__ void k_add_to_bf16(const float* __restrict__ a, const float* __restrict__ b,
                              unsigned short* __restrict__ o, size_t n)
{
    size_t i = (size_t)blockIdx.x * blockDim.x + threadIdx.x;
    if (i < n) o[i] = f2bf(a[i] + b[i]);
}

// softmax over 16 contiguous values per row (NL*NP), in place
__global__ void k_softmax16(float* __restrict__ aw, int rows)
{
    int i = blockIdx.x * blockDim.x + threadIdx.x;
    if (i >= rows) return;
    float* p = aw + (size_t)i * 16;
    float m = -1e30f;
#pragma unroll
    for (int j = 0; j < 16; ++j) m = fmaxf(m, p[j]);
    float e[16], s = 0.f;
#pragma unroll
    for (int j = 0; j < 16; ++j) { e[j] = __expf(p[j] - m); s += e[j]; }
    float inv = 1.f / s;
#pragma unroll
    for (int j = 0; j < 16; ++j) p[j] = e[j] * inv;
}

// ---------------------------------------------------------------------------
// Deformable bilinear sampling: one thread per (n,q,h,d). Gather-bound.
// value: (N, LEN, 256) f32 ; off: (3600,256) ; aw: (3600,128)
// ---------------------------------------------------------------------------
__global__ __launch_bounds__(256)
void k_deform_sample(const float* __restrict__ value,
                     const float* __restrict__ refp,
                     const float* __restrict__ off,
                     const float* __restrict__ aw,
                     unsigned short* __restrict__ outbf)
{
    int tid = blockIdx.x * blockDim.x + threadIdx.x;
    if (tid >= NB * LQ * 8 * 32) return;
    const int d = tid & 31;
    const int h = (tid >> 5) & 7;
    const int q = (tid >> 8) % LQ;
    const int n = tid / (256 * LQ);
    const int row = n * LQ + q;

    const int HH[4] = {100, 50, 25, 13};
    const int WW[4] = {134, 67, 34, 17};
    const int ST[4] = {0, 13400, 16750, 17600};

    float acc = 0.f;
#pragma unroll
    for (int l = 0; l < 4; ++l) {
        const float rx = refp[((size_t)row * 4 + l) * 2 + 0];
        const float ry = refp[((size_t)row * 4 + l) * 2 + 1];
        const int W = WW[l], H = HH[l];
        const float* vl = value + ((size_t)n * LEN + ST[l]) * 256 + h * 32 + d;
#pragma unroll
        for (int p = 0; p < 4; ++p) {
            float ox = off[(size_t)row * 256 + h * 32 + l * 8 + p * 2 + 0];
            float oy = off[(size_t)row * 256 + h * 32 + l * 8 + p * 2 + 1];
            float w  = aw[(size_t)row * 128 + h * 16 + l * 4 + p];
            float gx = (rx + ox / (float)W) * (float)W - 0.5f;
            float gy = (ry + oy / (float)H) * (float)H - 0.5f;
            float x0f = floorf(gx), y0f = floorf(gy);
            int   x0 = (int)x0f, y0 = (int)y0f;
            float wx1 = gx - x0f, wx0 = 1.f - wx1;
            float wy1 = gy - y0f, wy0 = 1.f - wy1;
            float s = 0.f;
            if (x0 >= 0 && x0 < W && y0 >= 0 && y0 < H)
                s += wx0 * wy0 * vl[(size_t)(y0 * W + x0) * 256];
            if (x0 + 1 >= 0 && x0 + 1 < W && y0 >= 0 && y0 < H)
                s += wx1 * wy0 * vl[(size_t)(y0 * W + x0 + 1) * 256];
            if (x0 >= 0 && x0 < W && y0 + 1 >= 0 && y0 + 1 < H)
                s += wx0 * wy1 * vl[(size_t)((y0 + 1) * W + x0) * 256];
            if (x0 + 1 >= 0 && x0 + 1 < W && y0 + 1 >= 0 && y0 + 1 < H)
                s += wx1 * wy1 * vl[(size_t)((y0 + 1) * W + x0 + 1) * 256];
            acc += w * s;
        }
    }
    outbf[(size_t)row * 256 + h * 32 + d] = f2bf(acc);
}

// ---------------------------------------------------------------------------
// Host orchestration
// ---------------------------------------------------------------------------
static inline dim3 gemm_grid(int M, int N) { return dim3((M + 63) / 64, (N + 63) / 64); }

extern "C" void kernel_launch(void* const* d_in, const int* in_sizes, int n_in,
                              void* d_out, int out_size, void* d_ws, size_t ws_size,
                              hipStream_t stream)
{
    (void)in_sizes; (void)n_in; (void)out_size; (void)ws_size;
    const float* queries  = (const float*)d_in[0];
    const float* features = (const float*)d_in[1];
    const float* refp     = (const float*)d_in[2];
    const float* q_pos    = (const float*)d_in[3];
    // d_in[4], d_in[5]: spatial shapes / level starts (hardcoded constants)
    const float* in_proj_w  = (const float*)d_in[6];
    const float* in_proj_b  = (const float*)d_in[7];
    const float* mha_out_w  = (const float*)d_in[8];
    const float* mha_out_b  = (const float*)d_in[9];
    const float* value_w    = (const float*)d_in[10];
    const float* value_b    = (const float*)d_in[11];
    const float* off_w      = (const float*)d_in[12];
    const float* off_b      = (const float*)d_in[13];
    const float* attn_w     = (const float*)d_in[14];
    const float* attn_b     = (const float*)d_in[15];
    const float* outp_w     = (const float*)d_in[16];
    const float* outp_b     = (const float*)d_in[17];
    const float* ffn_w1     = (const float*)d_in[18];
    const float* ffn_b1     = (const float*)d_in[19];
    const float* ffn_w2     = (const float*)d_in[20];
    const float* ffn_b2     = (const float*)d_in[21];
    const float* ln1_g = (const float*)d_in[22]; const float* ln1_b = (const float*)d_in[23];
    const float* ln2_g = (const float*)d_in[24]; const float* ln2_b = (const float*)d_in[25];
    const float* ln3_g = (const float*)d_in[26]; const float* ln3_b = (const float*)d_in[27];

    // ---- workspace carving (deterministic) ----
    size_t cur = 0;
    auto alloc = [&](size_t bytes) -> void* {
        cur = (cur + 255) & ~(size_t)255;
        void* p = (char*)d_ws + cur;
        cur += bytes;
        return p;
    };
    // bf16 weights
    unsigned short* wbf_inproj = (unsigned short*)alloc((size_t)768 * 256 * 2);
    unsigned short* wbf_mhaout = (unsigned short*)alloc((size_t)256 * 256 * 2);
    unsigned short* wbf_value  = (unsigned short*)alloc((size_t)256 * 384 * 2);
    unsigned short* wbf_off    = (unsigned short*)alloc((size_t)256 * 256 * 2);
    unsigned short* wbf_attn   = (unsigned short*)alloc((size_t)128 * 256 * 2);
    unsigned short* wbf_outp   = (unsigned short*)alloc((size_t)256 * 256 * 2);
    unsigned short* wbf_ffn1   = (unsigned short*)alloc((size_t)1024 * 256 * 2);
    unsigned short* wbf_ffn2   = (unsigned short*)alloc((size_t)256 * 1024 * 2);
    // bf16 activations
    unsigned short* qk_bf   = (unsigned short*)alloc((size_t)M36 * 256 * 2);
    unsigned short* qry_bf  = (unsigned short*)alloc((size_t)M36 * 256 * 2);
    unsigned short* qkv_bf  = (unsigned short*)alloc((size_t)M36 * 768 * 2);
    unsigned short* o_bf    = (unsigned short*)alloc((size_t)M36 * 256 * 2);
    unsigned short* qpe_bf  = (unsigned short*)alloc((size_t)M36 * 256 * 2);
    unsigned short* feat_bf = (unsigned short*)alloc((size_t)MFEAT * 384 * 2);
    unsigned short* samp_bf = (unsigned short*)alloc((size_t)M36 * 256 * 2);
    unsigned short* x2_bf   = (unsigned short*)alloc((size_t)M36 * 256 * 2);
    unsigned short* h1_bf   = (unsigned short*)alloc((size_t)M36 * 1024 * 2);
    // f32 buffers
    float* sa_f  = (float*)alloc((size_t)M36 * 256 * 4);
    float* x_f   = (float*)alloc((size_t)M36 * 256 * 4);
    float* val_f = (float*)alloc((size_t)MFEAT * 256 * 4);
    float* off_f = (float*)alloc((size_t)M36 * 256 * 4);
    float* aw_f  = (float*)alloc((size_t)M36 * 128 * 4);
    float* ca_f  = (float*)alloc((size_t)M36 * 256 * 4);
    float* x2_f  = (float*)alloc((size_t)M36 * 256 * 4);
    float* f_f   = (float*)alloc((size_t)M36 * 256 * 4);

    auto cvt = [&](const float* src, unsigned short* dst, size_t n) {
        k_to_bf16<<<dim3((unsigned)((n + 255) / 256)), dim3(256), 0, stream>>>(src, dst, n);
    };

    // 0) weights -> bf16
    cvt(in_proj_w, wbf_inproj, (size_t)768 * 256);
    cvt(mha_out_w, wbf_mhaout, (size_t)256 * 256);
    cvt(value_w,   wbf_value,  (size_t)256 * 384);
    cvt(off_w,     wbf_off,    (size_t)256 * 256);
    cvt(attn_w,    wbf_attn,   (size_t)128 * 256);
    cvt(outp_w,    wbf_outp,   (size_t)256 * 256);
    cvt(ffn_w1,    wbf_ffn1,   (size_t)1024 * 256);
    cvt(ffn_w2,    wbf_ffn2,   (size_t)256 * 1024);

    // 1) qk = queries + q_pos (bf16); queries (bf16); features (bf16)
    k_add_to_bf16<<<dim3((M36 * 256 + 255) / 256), dim3(256), 0, stream>>>(queries, q_pos, qk_bf, (size_t)M36 * 256);
    cvt(queries, qry_bf, (size_t)M36 * 256);
    cvt(features, feat_bf, (size_t)MFEAT * 384);

    // 2) QKV projections -> qkv_bf (3600 x 768)
    gemm_bf16_wmma<<<gemm_grid(M36, 512), dim3(128), 0, stream>>>(
        qk_bf, wbf_inproj, in_proj_b, nullptr, qkv_bf, M36, 512, 256, 768, 0);
    gemm_bf16_wmma<<<gemm_grid(M36, 256), dim3(128), 0, stream>>>(
        qry_bf, wbf_inproj + (size_t)512 * 256, in_proj_b + 512, nullptr, qkv_bf + 512,
        M36, 256, 256, 768, 0);

    // 3) flash self-attention -> o_bf (3600 x 256)
    mha_flash_wmma<<<dim3((LQ + 15) / 16, 8, NB), dim3(32), 0, stream>>>(qkv_bf, o_bf);

    // 4) MHA out-proj -> sa_f ; 5) x = LN(queries + sa)
    gemm_bf16_wmma<<<gemm_grid(M36, 256), dim3(128), 0, stream>>>(
        o_bf, wbf_mhaout, mha_out_b, sa_f, nullptr, M36, 256, 256, 256, 0);
    residual_ln<<<dim3(M36), dim3(256), 0, stream>>>(queries, sa_f, ln1_g, ln1_b, x_f, nullptr);

    // 6) qpe = x + q_pos (bf16)
    k_add_to_bf16<<<dim3((M36 * 256 + 255) / 256), dim3(256), 0, stream>>>(x_f, q_pos, qpe_bf, (size_t)M36 * 256);

    // 7) value projection (dominant GEMM): (71284 x 384) x (384 x 256)
    gemm_bf16_wmma<<<gemm_grid(MFEAT, 256), dim3(128), 0, stream>>>(
        feat_bf, wbf_value, value_b, val_f, nullptr, MFEAT, 256, 384, 256, 0);

    // 8) sampling offsets + attention weights
    gemm_bf16_wmma<<<gemm_grid(M36, 256), dim3(128), 0, stream>>>(
        qpe_bf, wbf_off, off_b, off_f, nullptr, M36, 256, 256, 256, 0);
    gemm_bf16_wmma<<<gemm_grid(M36, 128), dim3(128), 0, stream>>>(
        qpe_bf, wbf_attn, attn_b, aw_f, nullptr, M36, 128, 256, 128, 0);
    k_softmax16<<<dim3((M36 * 8 + 255) / 256), dim3(256), 0, stream>>>(aw_f, M36 * 8);

    // 9) deformable bilinear sampling -> samp_bf
    k_deform_sample<<<dim3((NB * LQ * 8 * 32 + 255) / 256), dim3(256), 0, stream>>>(
        val_f, refp, off_f, aw_f, samp_bf);

    // 10) out-proj -> ca ; x2 = LN(x + ca)
    gemm_bf16_wmma<<<gemm_grid(M36, 256), dim3(128), 0, stream>>>(
        samp_bf, wbf_outp, outp_b, ca_f, nullptr, M36, 256, 256, 256, 0);
    residual_ln<<<dim3(M36), dim3(256), 0, stream>>>(x_f, ca_f, ln2_g, ln2_b, x2_f, x2_bf);

    // 11) FFN: relu(x2 @ W1^T + b1) @ W2^T + b2 ; out = LN(x2 + f) -> d_out
    gemm_bf16_wmma<<<gemm_grid(M36, 1024), dim3(128), 0, stream>>>(
        x2_bf, wbf_ffn1, ffn_b1, nullptr, h1_bf, M36, 1024, 256, 1024, 1);
    gemm_bf16_wmma<<<gemm_grid(M36, 256), dim3(128), 0, stream>>>(
        h1_bf, wbf_ffn2, ffn_b2, f_f, nullptr, M36, 256, 1024, 256, 0);
    residual_ln<<<dim3(M36), dim3(256), 0, stream>>>(x2_f, f_f, ln3_g, ln3_b, (float*)d_out, nullptr);
}
